// DConv_89060441850475
// MI455X (gfx1250) — matrix operand order repaired and checked by reading, not compiled, not run on hardware
//
#include <hip/hip_runtime.h>
#include <math.h>

// ---------------- problem constants (from reference) ----------------
constexpr int Bb  = 8;
constexpr int C1c = 256;
constexpr int C2c = 256;
constexpr int Hh  = 64;
constexpr int Ww  = 64;
constexpr int GCc = 8;
constexpr int Cc  = 256;           // C1*1//GC*GC
constexpr int Gg  = Cc / GCc;      // 32
constexpr int Kk  = 9;
constexpr int OMom = 864;          // (G*K*3+7)//8*8 == G*K*3
constexpr int HWhw = Hh * Ww;      // 4096
constexpr int NPIX = Bb * HWhw;    // 32768
constexpr int KD   = 256;          // inner dim of every GEMM in this block
constexpr float EPSf = 1e-5f;

typedef __bf16 bf16;
typedef __attribute__((ext_vector_type(16))) __bf16 v16bf;
typedef __attribute__((ext_vector_type(8)))  float  v8f;
typedef int vint4 __attribute__((vector_size(16)));   // matches builtin param type

// Async global->LDS copy if the toolchain exposes it; otherwise ld+ds_store.
#if defined(__has_builtin)
#if __has_builtin(__builtin_amdgcn_global_load_async_to_lds_b128)
#define USE_ASYNC_LDS 1
#endif
#endif
#ifndef USE_ASYNC_LDS
#define USE_ASYNC_LDS 0
#endif

// ---------------- WMMA fragment loaders (bf16, wave32) ----------------
// A: 16x32 (MxK). lanes 0-15: M=lane, V0-3 = K0..7, V4-7 = K16..23
//                 lanes 16-31: M=lane-16, V0-3 = K8..15, V4-7 = K24..31
__device__ inline v16bf load_fragA_lds(const bf16* __restrict__ As, int m, int k0) {
    int lane = threadIdx.x & 31;
    int half = lane >> 4;
    int r    = lane & 15;
    const bf16* p = As + (m + r) * KD + k0 + half * 8;
    union { v16bf v; uint4 q[2]; } f;
    f.q[0] = *(const uint4*)(p);        // K +0..7   (8 bf16 = 16B)
    f.q[1] = *(const uint4*)(p + 16);   // K +16..23
    return f.v;
}

// B: 32x16 (KxN), stored as Wt row-major [N][Kd] so B[k][n] = Wt[n][k].
// lanes 0-15: n=lane, V0-3 = K0..7, V4-7 = K16..23 ; lanes 16-31: n=lane-16, V0-3 = K8..15, V4-7 = K24..31
__device__ inline v16bf load_fragB(const bf16* __restrict__ Wt, int n0, int k0) {
    int lane = threadIdx.x & 31;
    int half = lane >> 4;
    int r    = lane & 15;
    const bf16* p = Wt + (size_t)(n0 + r) * KD + k0 + half * 8;
    union { v16bf v; uint4 q[2]; } f;
    f.q[0] = *(const uint4*)(p);        // K +0..7
    f.q[1] = *(const uint4*)(p + 16);   // K +16..23
    return f.v;
}

// ---------------- GEMM: out[M,N] = A[M,256] * Wt[N,256]^T (+epilogue) ----------------
// Block: 256 threads = 8 waves. Block tile 64M x 128N; wave tile 64M x 16N.
// A strip (64x256 bf16 = 32KB) staged once in LDS, coalesced, shared by all waves.
// EPI 0: +bias, store bf16 row-major
// EPI 1: +bias, store f32 row-major
// EPI 2: BN affine + SiLU, scatter-store f32 to NCHW d_out
template<int EPI>
__global__ __launch_bounds__(256) void gemm_bf16_wmma(
        const bf16* __restrict__ A,
        const bf16* __restrict__ Wt,
        const float* __restrict__ bias,
        void* __restrict__ out,
        int M, int N,
        const float* __restrict__ bn_g,
        const float* __restrict__ bn_b,
        const float* __restrict__ bn_m,
        const float* __restrict__ bn_v) {
    __shared__ bf16 As[64 * KD];        // 32 KB of the 320 KB WGP LDS

    int m0 = blockIdx.x * 64;

    // ---- cooperative coalesced stage of the 64x256 A strip ----
    // 64*256*2B = 32768 B = 2048 x 16B chunks; 256 threads x 8 iterations.
    #pragma unroll
    for (int it = 0; it < 8; ++it) {
        int ci  = it * 256 + threadIdx.x;   // chunk id
        int row = ci >> 5;                  // 32 x 16B chunks per 512B row
        int cb  = (ci & 31) * 8;            // bf16 offset within row
        const bf16* gsrc = A + (size_t)(m0 + row) * KD + cb;
        bf16*       ldst = &As[row * KD + cb];
#if USE_ASYNC_LDS
        __builtin_amdgcn_global_load_async_to_lds_b128(
            (vint4*)const_cast<bf16*>(gsrc), (vint4*)ldst, 0, 0);
#else
        *(uint4*)ldst = *(const uint4*)gsrc;
#endif
    }
#if USE_ASYNC_LDS
    asm volatile("s_wait_asynccnt 0x0" ::: "memory");
#endif
    __syncthreads();

    int wave = threadIdx.x >> 5;
    int n0   = (blockIdx.y * 8 + wave) * 16;
    if (n0 >= N) return;                    // tail waves (N=864) exit after barrier

    v8f acc0 = {}, acc1 = {}, acc2 = {}, acc3 = {};
    for (int k0 = 0; k0 < KD; k0 += 32) {
        v16bf bf = load_fragB(Wt, n0, k0);
        v16bf a0 = load_fragA_lds(As,  0, k0);
        v16bf a1 = load_fragA_lds(As, 16, k0);
        v16bf a2 = load_fragA_lds(As, 32, k0);
        v16bf a3 = load_fragA_lds(As, 48, k0);
        acc0 = __builtin_amdgcn_wmma_f32_16x16x32_bf16(false, a0, false, bf, (short)0, acc0, false, false);
        acc1 = __builtin_amdgcn_wmma_f32_16x16x32_bf16(false, a1, false, bf, (short)0, acc1, false, false);
        acc2 = __builtin_amdgcn_wmma_f32_16x16x32_bf16(false, a2, false, bf, (short)0, acc2, false, false);
        acc3 = __builtin_amdgcn_wmma_f32_16x16x32_bf16(false, a3, false, bf, (short)0, acc3, false, false);
    }

    int lane = threadIdx.x & 31;
    int half = lane >> 4;
    int r    = lane & 15;
    int col  = n0 + r;

    float b = 0.f, s = 1.f, sh = 0.f;
    if constexpr (EPI == 0 || EPI == 1) {
        b = bias[col];
    } else {
        s  = bn_g[col] * rsqrtf(bn_v[col] + EPSf);
        sh = bn_b[col] - bn_m[col] * s;
    }

    v8f accs[4] = {acc0, acc1, acc2, acc3};
    #pragma unroll
    for (int t = 0; t < 4; ++t) {
        #pragma unroll
        for (int j = 0; j < 8; ++j) {
            int row = m0 + t * 16 + half * 8 + j;   // C/D layout: lanes 0-15 -> M=j, 16-31 -> M=j+8
            float v = accs[t][j];
            if constexpr (EPI == 0) {
                ((bf16*)out)[(size_t)row * N + col] = (bf16)(v + b);
            } else if constexpr (EPI == 1) {
                ((float*)out)[(size_t)row * N + col] = v + b;
            } else {
                float val = v * s + sh;
                float sig = 1.0f / (1.0f + __expf(-val));
                int n  = row / HWhw;
                int hw = row - n * HWhw;
                ((float*)out)[((size_t)n * C2c + col) * HWhw + hw] = val * sig;
            }
        }
    }
}

// ---------------- layout / precision packing ----------------
// x: f32 NCHW -> bf16 pixel-major [n*HW+hw][C1]
__global__ void pack_x_kernel(const float* __restrict__ x, bf16* __restrict__ xb) {
    size_t idx = (size_t)blockIdx.x * blockDim.x + threadIdx.x;
    if (idx >= (size_t)Bb * C1c * HWhw) return;
    int hw = idx % HWhw;
    int c  = (idx / HWhw) % C1c;
    int n  = idx / ((size_t)HWhw * C1c);
    xb[((size_t)n * HWhw + hw) * C1c + c] = (bf16)x[idx];
}

__global__ void pack_w_kernel(const float* __restrict__ src, bf16* __restrict__ dst, int count) {
    int i = blockIdx.x * blockDim.x + threadIdx.x;
    if (i < count) dst[i] = (bf16)src[i];
}

// fused weight: fw = cv2_w @ op_w  (256x256), stored bf16 row-major [C2][C]
__global__ void fuse_w_kernel(const float* __restrict__ cv2_w, const float* __restrict__ op_w,
                              bf16* __restrict__ fwb) {
    int idx = blockIdx.x * blockDim.x + threadIdx.x;
    if (idx >= C2c * Cc) return;
    int o = idx / Cc, c = idx % Cc;
    float acc = 0.f;
    for (int j = 0; j < Cc; ++j) acc += cv2_w[o * Cc + j] * op_w[j * Cc + c];
    fwb[idx] = (bf16)acc;
}

// ---------------- depthwise 3x3 'SAME' conv on pixel-major bf16 ----------------
__global__ void dwconv_kernel(const bf16* __restrict__ yb, const float* __restrict__ dw_w,
                              const float* __restrict__ dw_b, bf16* __restrict__ outb) {
    size_t idx = (size_t)blockIdx.x * blockDim.x + threadIdx.x;   // = p*C + c
    if (idx >= (size_t)NPIX * Cc) return;
    int c = idx & (Cc - 1);
    int p = idx >> 8;
    int hw = p % HWhw;
    int n  = p / HWhw;
    int h  = hw / Ww, w = hw % Ww;
    float acc = dw_b[c];
    #pragma unroll
    for (int dy = -1; dy <= 1; ++dy) {
        #pragma unroll
        for (int dx = -1; dx <= 1; ++dx) {
            int y = h + dy, x = w + dx;
            if (y >= 0 && y < Hh && x >= 0 && x < Ww) {
                float v = (float)yb[((size_t)n * HWhw + y * Ww + x) * Cc + c];
                acc += v * dw_w[c * 9 + (dy + 1) * 3 + (dx + 1)];
            }
        }
    }
    outb[idx] = (bf16)acc;
}

// ---------------- deformable gather core ----------------
// thread per (pixel, group); per-group 27-vector of om: [2k]=dy off, [2k+1]=dx off, [18+k]=mask
__global__ void dcn_kernel(const bf16* __restrict__ vb, const float* __restrict__ om,
                           bf16* __restrict__ db) {
    size_t idx = (size_t)blockIdx.x * blockDim.x + threadIdx.x;
    if (idx >= (size_t)NPIX * Gg) return;
    int g = idx % Gg;
    int p = idx / Gg;
    int hw = p % HWhw;
    int n  = p / HWhw;
    int h  = hw / Ww, w = hw % Ww;
    const float* o27 = om + (size_t)p * OMom + g * 27;

    float res[GCc];
    #pragma unroll
    for (int c = 0; c < GCc; ++c) res[c] = 0.f;

    #pragma unroll
    for (int k = 0; k < Kk; ++k) {
        float offy = o27[2 * k];
        float offx = o27[2 * k + 1];
        float mk   = o27[18 + k];
        float ys = (float)h + (float)(k / 3 - 1) + offy;
        float xs = (float)w + (float)(k % 3 - 1) + offx;
        float y0f = floorf(ys), x0f = floorf(xs);
        int y0 = (int)y0f, x0 = (int)x0f;
        float fy = ys - y0f, fx = xs - x0f;
        #pragma unroll
        for (int oy = 0; oy < 2; ++oy) {
            #pragma unroll
            for (int ox = 0; ox < 2; ++ox) {
                int yy = y0 + oy, xx = x0 + ox;
                if (yy >= 0 && yy < Hh && xx >= 0 && xx < Ww) {
                    float wgt = (oy ? fy : 1.f - fy) * (ox ? fx : 1.f - fx) * mk;
                    const bf16* vp = vb + ((size_t)n * HWhw + yy * Ww + xx) * Cc + g * GCc;
                    #pragma unroll
                    for (int c = 0; c < GCc; ++c) res[c] += wgt * (float)vp[c];
                }
            }
        }
    }
    bf16* dp = db + (size_t)p * Cc + g * GCc;
    #pragma unroll
    for (int c = 0; c < GCc; ++c) dp[c] = (bf16)res[c];
}

// ---------------- host launch ----------------
extern "C" void kernel_launch(void* const* d_in, const int* in_sizes, int n_in,
                              void* d_out, int out_size, void* d_ws, size_t ws_size,
                              hipStream_t stream) {
    const float* x      = (const float*)d_in[0];
    const float* cv1_w  = (const float*)d_in[1];
    const float* cv1_b  = (const float*)d_in[2];
    const float* vp_w   = (const float*)d_in[3];
    const float* vp_b   = (const float*)d_in[4];
    const float* dw_w   = (const float*)d_in[5];
    const float* dw_b   = (const float*)d_in[6];
    const float* om_w   = (const float*)d_in[7];
    const float* om_b   = (const float*)d_in[8];
    const float* op_w   = (const float*)d_in[9];
    const float* cv2_w  = (const float*)d_in[10];
    const float* bn_g   = (const float*)d_in[11];
    const float* bn_b   = (const float*)d_in[12];
    const float* bn_m   = (const float*)d_in[13];
    const float* bn_v   = (const float*)d_in[14];
    float* out = (float*)d_out;

    // workspace carve-up (256B aligned)
    char* wp = (char*)d_ws;
    auto alloc = [&](size_t bytes) -> void* {
        void* p = (void*)wp;
        wp += (bytes + 255) & ~(size_t)255;
        return p;
    };
    bf16*  xb    = (bf16*)alloc((size_t)NPIX * C1c * sizeof(bf16));
    bf16*  yb    = (bf16*)alloc((size_t)NPIX * Cc  * sizeof(bf16));
    bf16*  vb    = (bf16*)alloc((size_t)NPIX * Cc  * sizeof(bf16));
    bf16*  dwb   = (bf16*)alloc((size_t)NPIX * Cc  * sizeof(bf16));
    bf16*  db    = (bf16*)alloc((size_t)NPIX * Cc  * sizeof(bf16));
    float* omf   = (float*)alloc((size_t)NPIX * OMom * sizeof(float));
    bf16*  wcv1b = (bf16*)alloc((size_t)Cc * C1c * sizeof(bf16));
    bf16*  wvpb  = (bf16*)alloc((size_t)Cc * Cc  * sizeof(bf16));
    bf16*  womb  = (bf16*)alloc((size_t)OMom * Cc * sizeof(bf16));
    bf16*  fwb   = (bf16*)alloc((size_t)C2c * Cc * sizeof(bf16));

    const int TB = 256;

    // 1) precision/layout packs
    {
        size_t n = (size_t)Bb * C1c * HWhw;
        pack_x_kernel<<<dim3((n + TB - 1) / TB), dim3(TB), 0, stream>>>(x, xb);
    }
    pack_w_kernel<<<dim3((Cc * C1c + TB - 1) / TB), dim3(TB), 0, stream>>>(cv1_w, wcv1b, Cc * C1c);
    pack_w_kernel<<<dim3((Cc * Cc + TB - 1) / TB), dim3(TB), 0, stream>>>(vp_w, wvpb, Cc * Cc);
    pack_w_kernel<<<dim3((OMom * Cc + TB - 1) / TB), dim3(TB), 0, stream>>>(om_w, womb, OMom * Cc);
    fuse_w_kernel<<<dim3((C2c * Cc + TB - 1) / TB), dim3(TB), 0, stream>>>(cv2_w, op_w, fwb);

    // 2) cv1: y = x @ cv1_w^T + b   (NPIX x 256)
    gemm_bf16_wmma<0><<<dim3(NPIX / 64, (Cc + 127) / 128), dim3(TB), 0, stream>>>(
        xb, wcv1b, cv1_b, yb, NPIX, Cc, nullptr, nullptr, nullptr, nullptr);

    // 3) vp: v = y @ vp_w^T + b
    gemm_bf16_wmma<0><<<dim3(NPIX / 64, (Cc + 127) / 128), dim3(TB), 0, stream>>>(
        yb, wvpb, vp_b, vb, NPIX, Cc, nullptr, nullptr, nullptr, nullptr);

    // 4) depthwise 3x3 on y
    {
        size_t n = (size_t)NPIX * Cc;
        dwconv_kernel<<<dim3((n + TB - 1) / TB), dim3(TB), 0, stream>>>(yb, dw_w, dw_b, dwb);
    }

    // 5) om head: om = dw @ om_w^T + b   (NPIX x 864, f32)
    gemm_bf16_wmma<1><<<dim3(NPIX / 64, (OMom + 127) / 128), dim3(TB), 0, stream>>>(
        dwb, womb, om_b, omf, NPIX, OMom, nullptr, nullptr, nullptr, nullptr);

    // 6) deformable bilinear gather
    {
        size_t n = (size_t)NPIX * Gg;
        dcn_kernel<<<dim3((n + TB - 1) / TB), dim3(TB), 0, stream>>>(vb, omf, db);
    }

    // 7) fused (op @ cv2) GEMM + BN + SiLU + NCHW scatter to d_out
    gemm_bf16_wmma<2><<<dim3(NPIX / 64, (C2c + 127) / 128), dim3(TB), 0, stream>>>(
        db, fwb, nullptr, out, NPIX, C2c, bn_g, bn_b, bn_m, bn_v);
}
